// NeuralNetwork_31447750541324
// MI455X (gfx1250) — matrix operand order repaired and checked by reading, not compiled, not run on hardware
//
#include <hip/hip_runtime.h>
#include <hip/hip_bf16.h>

typedef __attribute__((ext_vector_type(16))) _Float16 v16h;
typedef __attribute__((ext_vector_type(8)))  _Float16 v8h;
typedef __attribute__((ext_vector_type(8)))  float    v8f;

__device__ __forceinline__ v8f wmma_f16(v16h a, v16h b, v8f c) {
    // v_wmma_f32_16x16x32_f16 : D = A(16x32 f16) x B(32x16 f16) + C(16x16 f32)
    return __builtin_amdgcn_wmma_f32_16x16x32_f16(false, a, false, b, (short)0, c, false, false);
}

// ---------------------------------------------------------------------------
// Stage 0: convert W1 [64][3][9][9] f32 -> f16 [n][k] with K padded 243->256
// ---------------------------------------------------------------------------
__global__ void k_prep_w1(const float* __restrict__ W1, _Float16* __restrict__ w1h) {
    int i = blockIdx.x * blockDim.x + threadIdx.x;
    if (i < 64 * 256) {
        int n = i >> 8, k = i & 255;
        float v = (k < 243) ? W1[n * 243 + k] : 0.f;
        w1h[i] = (_Float16)v;
    }
}

// ---------------------------------------------------------------------------
// Stage 1: conv1 (stride==kernel==9 -> non-overlapping patches == pure GEMM)
//          fused bias + ReLU + 2x2 maxpool epilogue through LDS.
// grid = 640 frames * 5 row-blocks, block = 320 (10 waves)
// Workgroup tile: M=80 output positions (4 rows x 20 cols), N=64 channels.
// Wave w: mi = w%5 (16-row M subtile), covers N columns [ (w/5)*32, +32 ).
// The im2col fill exploits blockDim%32==0: the K lane (kk = tid&31) is
// invariant across the fill loop, so the k->(c,ky,kx) decomposition hoists
// to once per chunk, and ml = (tid>>5) + 10*j gives closed-form (ry, ox).
// ---------------------------------------------------------------------------
__global__ __launch_bounds__(320) void k_conv1(const float* __restrict__ x,
                                               const _Float16* __restrict__ w1h,
                                               const float* __restrict__ b1,
                                               float* __restrict__ pooled) {
    __shared__ _Float16 a_lds[80 * 32];   // A chunk (M x K32), f16
    __shared__ float    c_lds[80 * 64];   // C tile for pooled epilogue

    const int f    = blockIdx.x / 5;
    const int oyb  = blockIdx.x % 5;      // 4 conv-output rows per block
    const int tid  = threadIdx.x;
    const int lane = tid & 31;
    const int wv   = tid >> 5;            // 0..9
    const int mi   = wv % 5;
    const int ni2  = wv / 5;              // 0..1 -> 32-wide N slice
    const int hl   = lane & 15;
    const int khalf = (lane < 16) ? 0 : 1;

    const int kk  = tid & 31;             // K lane within chunk (fill role)
    const int mlb = tid >> 5;             // 0..9 base output position (fill role)

    const float* xf = x + (size_t)f * (3 * 180 * 180);

    v8f acc0 = {}; v8f acc1 = {};

    for (int kc = 0; kc < 8; ++kc) {      // K = 243 padded to 256, chunks of 32
        __syncthreads();                  // previous chunk's fragments consumed

        // ---- hoisted per-chunk index math (once, not per element) ----
        int k = kc * 32 + kk;
        bool valid = (k < 243);
        int c  = k / 81;
        int t  = k - c * 81;
        int ky = t / 9;
        int kx = t - ky * 9;
        // clamp address for pad lanes (branchless, no OOB speculation)
        int goff = valid ? ((c * 180 + ky) * 180 + kx) : 0;
        const float* gp = xf + goff + (size_t)(oyb * 4) * 1620; // + ry*1620 + ox*9
        _Float16* lp = &a_lds[mlb * 32 + kk];                   // + 320*j

        #pragma unroll
        for (int j = 0; j < 8; ++j) {
            int ry = j >> 1;
            int ox = mlb + ((j & 1) ? 10 : 0);
            float v = gp[ry * 1620 + ox * 9];
            lp[320 * j] = (_Float16)(valid ? v : 0.f);
        }
        __syncthreads();

        // A fragment (contiguous 16-half K run per lane, starting at 16*(lane>=16))
        int m   = mi * 16 + hl;
        int kk0 = khalf * 16;
        v8h lo = *(const v8h*)&a_lds[m * 32 + kk0];
        v8h hi = *(const v8h*)&a_lds[m * 32 + kk0 + 8];
        v16h a;
        #pragma unroll
        for (int j = 0; j < 8; ++j) { a[j] = lo[j]; a[j + 8] = hi[j]; }

        // B fragments straight from the L2-resident f16 weights (contiguous
        // 16-half run per lane: column n, K-run starts at kc*32 + 16*(lane>=16))
        int k0 = kc * 32 + khalf * 16;
        int n0 = (ni2 * 2 + 0) * 16 + hl;
        int n1 = (ni2 * 2 + 1) * 16 + hl;
        v16h b0  = *(const v16h*)(w1h + n0 * 256 + k0);
        v16h b1v = *(const v16h*)(w1h + n1 * 256 + k0);

        acc0 = wmma_f16(a, b0, acc0);
        acc1 = wmma_f16(a, b1v, acc1);
    }

    // Spill C tile to LDS (C layout: VGPR v -> M = v + 8*(lane>=16), N = lane%16)
    {
        int base_m = mi * 16 + ((lane < 16) ? 0 : 8);
        int n0 = (ni2 * 2 + 0) * 16 + hl;
        int n1 = (ni2 * 2 + 1) * 16 + hl;
        #pragma unroll
        for (int vv = 0; vv < 8; ++vv) {
            c_lds[(base_m + vv) * 64 + n0] = acc0[vv];
            c_lds[(base_m + vv) * 64 + n1] = acc1[vv];
        }
    }
    __syncthreads();

    // Fused bias + ReLU + 2x2 maxpool: 64ch x 2 pooled rows x 10 pooled cols
    for (int o = tid; o < 64 * 2 * 10; o += 320) {
        int ch = o / 20;
        int rr = o - ch * 20;
        int py = rr / 10;
        int px = rr - py * 10;
        float bb = b1[ch];
        float mx = 0.f;                   // ReLU floor
        #pragma unroll
        for (int dy = 0; dy < 2; ++dy)
            #pragma unroll
            for (int dx = 0; dx < 2; ++dx) {
                int ml = (2 * py + dy) * 20 + 2 * px + dx;
                mx = fmaxf(mx, c_lds[ml * 64 + ch] + bb);
            }
        pooled[((size_t)(f * 64 + ch) * 10 + (oyb * 2 + py)) * 10 + px] = mx;
    }
}

// ---------------------------------------------------------------------------
// Stage 2: conv2 (stride==kernel==5), bias+ReLU+2x2 pool -> y[640][3]
// grid = 640 frames, block = 384 (12 waves: one wave per (och, pos))
// ---------------------------------------------------------------------------
__global__ __launch_bounds__(384) void k_conv2(const float* __restrict__ pooled,
                                               const float* __restrict__ W2,
                                               const float* __restrict__ b2,
                                               float* __restrict__ y) {
    __shared__ float s[12];
    const int f    = blockIdx.x;
    const int tid  = threadIdx.x;
    const int lane = tid & 31;
    const int g    = tid >> 5;            // 0..11
    const int och  = g >> 2;
    const int pos  = g & 3;
    const int py   = pos >> 1, px = pos & 1;

    float sum = 0.f;
    for (int k = lane; k < 1600; k += 32) {     // K = 64ch * 5 * 5
        int c  = k / 25;
        int t  = k - c * 25;
        int ky = t / 5;
        int kx = t - ky * 5;
        float a = pooled[((size_t)(f * 64 + c) * 10 + py * 5 + ky) * 10 + px * 5 + kx];
        sum += a * W2[och * 1600 + k];
    }
    #pragma unroll
    for (int off = 16; off > 0; off >>= 1) sum += __shfl_down(sum, off, 32);
    if (lane == 0) s[g] = sum;
    __syncthreads();
    if (tid < 3) {
        float bb = b2[tid];
        float mx = 0.f;
        #pragma unroll
        for (int p = 0; p < 4; ++p) mx = fmaxf(mx, s[tid * 4 + p] + bb);
        y[f * 3 + tid] = mx;
    }
}

// ---------------------------------------------------------------------------
// Stage 3: BatchNorm1d(F) over (B,3) + per-sample stable argsort of channel
//          ranges + build RNN input  seq[3][64][16] (B padded 10->16 w/ zeros)
// single workgroup, 192 threads
// ---------------------------------------------------------------------------
__global__ __launch_bounds__(192) void k_bn_reorder(const float* __restrict__ y,
                                                    const float* __restrict__ gamma,
                                                    const float* __restrict__ beta,
                                                    float* __restrict__ seq) {
    __shared__ float ybn[10 * 64 * 3];
    __shared__ float rng[30];
    __shared__ int   perm[30];
    const int tid = threadIdx.x;

    if (tid < 64) {
        float s = 0.f, s2 = 0.f;
        for (int b = 0; b < 10; ++b)
            for (int c = 0; c < 3; ++c) {
                float v = y[(b * 64 + tid) * 3 + c];
                s += v; s2 += v * v;
            }
        float mean = s * (1.f / 30.f);
        float var  = s2 * (1.f / 30.f) - mean * mean;
        float inv  = rsqrtf(var + 1e-5f) * gamma[tid];
        float bt   = beta[tid];
        for (int b = 0; b < 10; ++b)
            for (int c = 0; c < 3; ++c) {
                float v = y[(b * 64 + tid) * 3 + c];
                ybn[(b * 64 + tid) * 3 + c] = (v - mean) * inv + bt;
            }
    }
    __syncthreads();
    if (tid < 30) {
        int b = tid / 3, c = tid - b * 3;
        float mn = 1e30f, mx = -1e30f;
        for (int ff = 0; ff < 64; ++ff) {
            float v = ybn[(b * 64 + ff) * 3 + c];
            mn = fminf(mn, v); mx = fmaxf(mx, v);
        }
        rng[tid] = mx - mn;
    }
    __syncthreads();
    if (tid < 10) {  // stable 3-element ascending argsort (matches jnp.argsort)
        float v0 = rng[tid*3+0], v1 = rng[tid*3+1], v2 = rng[tid*3+2];
        int i0 = 0, i1 = 1, i2 = 2;
        if (v1 < v0) { float tv=v0; v0=v1; v1=tv; int ti=i0; i0=i1; i1=ti; }
        if (v2 < v1) { float tv=v1; v1=v2; v2=tv; int ti=i1; i1=i2; i2=ti; }
        if (v1 < v0) { float tv=v0; v0=v1; v1=tv; int ti=i0; i0=i1; i1=ti; }
        perm[tid*3+0]=i0; perm[tid*3+1]=i1; perm[tid*3+2]=i2;
    }
    __syncthreads();
    for (int i = tid; i < 3 * 64 * 16; i += 192) {
        int r = i / 1024, rem = i - r * 1024;
        int t = rem >> 4, b = rem & 15;
        float v = 0.f;
        if (b < 10) v = ybn[(b * 64 + t) * 3 + perm[b * 3 + r]];
        seq[i] = v;
    }
}

// ---------------------------------------------------------------------------
// Stage 4: the 3 Elman RNNs, one workgroup each (16 waves, 512 threads).
// h (16x256) lives in LDS as f16; Whh B-fragments stay resident in VGPRs
// (8 frags x 8 VGPRs/wave). Each step: 8x v_wmma_f32_16x16x32_f16 + tanh.
// ---------------------------------------------------------------------------
__global__ __launch_bounds__(512) void k_rnn(const float* __restrict__ seq,
                                             const float* __restrict__ Wih,
                                             const float* __restrict__ Whh,
                                             const float* __restrict__ bih,
                                             const float* __restrict__ bhh,
                                             float* __restrict__ lasts) {
    __shared__ _Float16 h_lds[16 * 256];  // 8 KB
    __shared__ float    x_lds[64 * 16];   // 4 KB

    const int r    = blockIdx.x;
    const int tid  = threadIdx.x;
    const int lane = tid & 31;
    const int wi   = tid >> 5;            // N-tile 0..15
    const int hl   = lane & 15;
    const int n    = wi * 16 + hl;        // this lane's output column

    for (int i = tid; i < 64 * 16; i += 512) x_lds[i] = seq[r * 1024 + i];
    for (int i = tid; i < 16 * 256; i += 512) h_lds[i] = (_Float16)0.f;

    const float wihn = Wih[r * 256 + n];
    const float cn   = bih[r * 256 + n] + bhh[r * 256 + n];

    // Register-resident B fragments: B[k][n] = Whh[n][k]; per lane a
    // contiguous 16-half K run of column n starting at kc*32 + 16*(lane>=16).
    v16h bfr[8];
    {
        const float* wp = Whh + (size_t)(r * 256 + n) * 256;
        #pragma unroll
        for (int kc = 0; kc < 8; ++kc) {
            int k0 = kc * 32 + ((lane < 16) ? 0 : 16);
            v16h bb;
            #pragma unroll
            for (int j = 0; j < 16; ++j) bb[j] = (_Float16)wp[k0 + j];
            bfr[kc] = bb;
        }
    }
    __syncthreads();

    const int kkoff = (lane < 16) ? 0 : 16;
    const int boff  = (lane < 16) ? 0 : 8;

    for (int t = 0; t < 64; ++t) {
        v8f acc = {};
        #pragma unroll
        for (int kc = 0; kc < 8; ++kc) {
            int base = hl * 256 + kc * 32 + kkoff;
            v8h lo = *(const v8h*)&h_lds[base];
            v8h hi = *(const v8h*)&h_lds[base + 8];
            v16h a;
            #pragma unroll
            for (int j = 0; j < 8; ++j) { a[j] = lo[j]; a[j + 8] = hi[j]; }
            acc = wmma_f16(a, bfr[kc], acc);
        }
        __syncthreads();                  // everyone done reading old h
        #pragma unroll
        for (int vv = 0; vv < 8; ++vv) {
            int b = vv + boff;
            float hn = tanhf(acc[vv] + x_lds[t * 16 + b] * wihn + cn);
            h_lds[b * 256 + n] = (_Float16)hn;
            if (t == 63) lasts[(size_t)(r * 16 + b) * 256 + n] = hn;
        }
        __syncthreads();                  // new h visible
    }
}

// ---------------------------------------------------------------------------
// Stage 5: average the 3 last hiddens + linear head -> out[10][5]
// ---------------------------------------------------------------------------
__global__ void k_final(const float* __restrict__ lasts, const float* __restrict__ Wl,
                        const float* __restrict__ bl, float* __restrict__ out) {
    int tid = threadIdx.x;
    if (tid < 50) {
        int b = tid / 5, j = tid - b * 5;
        float s = 0.f;
        for (int nn = 0; nn < 256; ++nn) {
            float hv = lasts[b * 256 + nn] + lasts[4096 + b * 256 + nn]
                     + lasts[8192 + b * 256 + nn];
            s += hv * Wl[j * 256 + nn];
        }
        out[b * 5 + j] = s * (1.f / 3.f) + bl[j];
    }
}

// ---------------------------------------------------------------------------
extern "C" void kernel_launch(void* const* d_in, const int* in_sizes, int n_in,
                              void* d_out, int out_size, void* d_ws, size_t ws_size,
                              hipStream_t stream) {
    (void)in_sizes; (void)n_in; (void)out_size; (void)ws_size;
    const float* x     = (const float*)d_in[0];
    const float* W1    = (const float*)d_in[1];
    const float* b1    = (const float*)d_in[2];
    const float* W2    = (const float*)d_in[3];
    const float* b2    = (const float*)d_in[4];
    const float* gamma = (const float*)d_in[5];
    const float* beta  = (const float*)d_in[6];
    const float* Wih   = (const float*)d_in[7];
    const float* Whh   = (const float*)d_in[8];
    const float* bih   = (const float*)d_in[9];
    const float* bhh   = (const float*)d_in[10];
    const float* Wl    = (const float*)d_in[11];
    const float* bl    = (const float*)d_in[12];
    float* out = (float*)d_out;

    char* ws = (char*)d_ws;
    _Float16* w1h    = (_Float16*)ws;                                  // 32 KB
    float*    pooled = (float*)(ws + 32768);                           // 640*64*100 f32
    float*    y      = (float*)(ws + 32768 + 16384000);                // 640*3 f32
    float*    seq    = (float*)(ws + 32768 + 16384000 + 7680);         // 3*64*16 f32
    float*    lasts  = (float*)(ws + 32768 + 16384000 + 7680 + 12288); // 3*16*256 f32

    k_prep_w1   <<<64,   256, 0, stream>>>(W1, w1h);
    k_conv1     <<<3200, 320, 0, stream>>>(x, w1h, b1, pooled);
    k_conv2     <<<640,  384, 0, stream>>>(pooled, W2, b2, y);
    k_bn_reorder<<<1,    192, 0, stream>>>(y, gamma, beta, seq);
    k_rnn       <<<3,    512, 0, stream>>>(seq, Wih, Whh, bih, bhh, lasts);
    k_final     <<<1,     64, 0, stream>>>(lasts, Wl, bl, out);
}
